// LATTEConv_79448305041722
// MI455X (gfx1250) — matrix operand order repaired and checked by reading, not compiled, not run on hardware
//
#include <hip/hip_runtime.h>
#include <stdint.h>

// ---------------- problem constants (match reference) ----------------
#define N_NODES 50000
#define DIM     256
#define HEADS   4
#define HDIM    64
#define NEG     0.2f
#define LN_EPS  1e-5f

// ---------------- tiling ----------------
#define BM   64          // rows per workgroup
#define LDA  264         // padded bf16 row stride for the A (feat) LDS tile

typedef __attribute__((ext_vector_type(16))) __bf16 v16bf;
typedef __attribute__((ext_vector_type(8)))  __bf16 v8bf;
typedef __attribute__((ext_vector_type(8)))  float  v8f;
typedef __attribute__((ext_vector_type(4)))  float  v4f;

// ---------------- indicator kernels ----------------
__global__ void zero_ind_kernel(uint8_t* __restrict__ ind, int n) {
    int i = blockIdx.x * blockDim.x + threadIdx.x;
    if (i < n) ind[i] = 0;
}

__global__ void mark_kernel(const int* __restrict__ dst, uint8_t* __restrict__ ind, int e) {
    int i = blockIdx.x * blockDim.x + threadIdx.x;
    if (i < e) ind[dst[i]] = 1;   // plain store of constant; no atomic needed
}

// ---------------- Wr -> Wr^T in bf16 (one-time, L2-resident afterwards) ----------------
__global__ void wrT_kernel(const float* __restrict__ Wr, unsigned short* __restrict__ WrTu) {
    int k = blockIdx.x;      // input-dim index (K)
    int c = threadIdx.x;     // output column (N)
    ((__bf16*)WrTu)[c * DIM + k] = (__bf16)Wr[k * DIM + c];
}

// ---------------- fused: v = feat@Wr + br  -> metapath attention -> ReLU -> LayerNorm ----------------
__global__ __launch_bounds__(256)
void latte_fused(const float* __restrict__ feat,
                 const unsigned short* __restrict__ WrTu,
                 const float* __restrict__ br,
                 const uint8_t* __restrict__ ind,
                 const float* __restrict__ rel_q,
                 const float* __restrict__ rel_k,
                 const float* __restrict__ lng,
                 const float* __restrict__ lnb,
                 float* __restrict__ out) {
    // one 64KB LDS region, used first for the bf16 A tile, then reused for the fp32 V tile
    __shared__ __align__(16) unsigned char smem[65536];
    __bf16* Alds = (__bf16*)smem;            // [BM][LDA] bf16   (33792 B)
    float*  Vlds = (float*)smem;             // [BM][DIM] fp32   (65536 B)
    const __bf16* WrT = (const __bf16*)WrTu;

    const int tid  = threadIdx.x;
    const int wave = tid >> 5;
    const int lane = tid & 31;
    const int row0 = blockIdx.x * BM;

    // ---- stage A tile: feat[row0 .. row0+63][0..255] fp32 -> bf16 LDS (padded stride) ----
    {
        const int r    = tid >> 2;           // 0..63 local row
        const int seg  = (tid & 3) * 64;     // 64-col segment
        const int grow = row0 + r;
        #pragma unroll
        for (int i = 0; i < 16; ++i) {
            const int c = seg + i * 4;
            v4f f = {0.f, 0.f, 0.f, 0.f};
            if (grow < N_NODES) f = *(const v4f*)(feat + (size_t)grow * DIM + c);
            __bf16* p = Alds + r * LDA + c;
            p[0] = (__bf16)f.x; p[1] = (__bf16)f.y; p[2] = (__bf16)f.z; p[3] = (__bf16)f.w;
        }
    }
    __syncthreads();

    // wave -> (row tile, column half);  8 waves cover 4 row tiles x 256 cols
    const int rt       = wave & 3;           // 16-row tile
    const int colHalf  = wave >> 2;          // 0: cols 0..127, 1: cols 128..255
    const int m        = lane & 15;
    const int khalf    = lane >> 4;
    const int colBase  = colHalf * 128;

    v8f acc[8] = {};                         // 8 16x16 f32 accumulators per wave

    for (int kc = 0; kc < 8; ++kc) {         // K chunks of 32
        const int k0 = kc * 32;
        // A fragment (16x32 bf16): lanes 0-15 K=[k0..k0+7, k0+16..k0+23], lanes 16-31 +8
        const __bf16* ap = Alds + (rt * 16 + m) * LDA + k0 + khalf * 8;
        v8bf alo = *(const v8bf*)ap;
        v8bf ahi = *(const v8bf*)(ap + 16);
        v16bf afrag;
        #pragma unroll
        for (int i = 0; i < 8; ++i) { afrag[i] = alo[i]; afrag[8 + i] = ahi[i]; }

        #pragma unroll
        for (int ct = 0; ct < 8; ++ct) {
            const int col = colBase + ct * 16 + m;
            // B fragment (32x16 bf16): lanes 0-15 K=[k0..k0+7, k0+16..k0+23], lanes 16-31 +8, from Wr^T row `col`
            const __bf16* bp = WrT + (size_t)col * DIM + k0 + khalf * 8;
            v8bf blo = *(const v8bf*)bp;
            v8bf bhi = *(const v8bf*)(bp + 16);
            v16bf bfrag;
            #pragma unroll
            for (int i = 0; i < 8; ++i) { bfrag[i] = blo[i]; bfrag[8 + i] = bhi[i]; }
            acc[ct] = __builtin_amdgcn_wmma_f32_16x16x32_bf16(
                false, afrag, false, bfrag, (short)0, acc[ct], false, false);
        }
    }

    __syncthreads();   // all waves done reading A; LDS now becomes the V tile

    // scatter accumulators (+ bias) into V tile. C/D layout: VGPR j -> row j (+8 for hi lanes)
    #pragma unroll
    for (int ct = 0; ct < 8; ++ct) {
        const int col = colBase + ct * 16 + m;
        const float b = br[col];
        #pragma unroll
        for (int j = 0; j < 8; ++j) {
            const int rloc = rt * 16 + khalf * 8 + j;
            Vlds[rloc * DIM + col] = acc[ct][j] + b;
        }
    }
    __syncthreads();

    // ---- fused epilogue: per-row metapath attention + ReLU + LayerNorm ----
    const int h    = lane >> 3;              // head for this lane's 8 columns
    const int cgrp = (lane & 7) * 8;         // within-head column base
    for (int rr = 0; rr < 8; ++rr) {
        const int rloc = wave * 8 + rr;
        const int n    = row0 + rloc;
        const float* vrow = Vlds + rloc * DIM;

        float v[8];
        #pragma unroll
        for (int i = 0; i < 8; ++i) v[i] = vrow[lane * 8 + i];

        // q = v . rel_q, kv = v . rel_k per head (8-lane segmented butterfly reduce)
        float qp = 0.f, kp = 0.f;
        #pragma unroll
        for (int i = 0; i < 8; ++i) {
            qp += v[i] * rel_q[h * HDIM + cgrp + i];
            kp += v[i] * rel_k[h * HDIM + cgrp + i];
        }
        #pragma unroll
        for (int s = 1; s < 8; s <<= 1) {
            qp += __shfl_xor(qp, s, 32);
            kp += __shfl_xor(kp, s, 32);
        }

        float i0 = 0.f, i1 = 0.f, i2 = 0.f;
        if (n < N_NODES) {
            i0 = (float)ind[n];
            i1 = (float)ind[N_NODES + n];
            i2 = (float)ind[2 * N_NODES + n];
        }

        // beta = softmax_t( lrelu(q + ind_t * kv) ), t=0..2 relations + self
        auto lrelu = [](float z) { return z > 0.f ? z : NEG * z; };
        const float z0 = lrelu(qp + i0 * kp);
        const float z1 = lrelu(qp + i1 * kp);
        const float z2 = lrelu(qp + i2 * kp);
        const float z3 = lrelu(qp + kp);
        const float mz = fmaxf(fmaxf(z0, z1), fmaxf(z2, z3));
        const float e0 = __expf(z0 - mz), e1 = __expf(z1 - mz);
        const float e2 = __expf(z2 - mz), e3 = __expf(z3 - mz);
        const float es = e0 + e1 + e2 + e3;
        const float w  = (e0 * i0 + e1 * i1 + e2 * i2 + e3) / es;

        // out = LN(relu(v * w))
        float y[8]; float s1 = 0.f, s2 = 0.f;
        #pragma unroll
        for (int i = 0; i < 8; ++i) {
            float t = v[i] * w;
            t = t > 0.f ? t : 0.f;
            y[i] = t; s1 += t; s2 += t * t;
        }
        #pragma unroll
        for (int s = 1; s < 32; s <<= 1) {
            s1 += __shfl_xor(s1, s, 32);
            s2 += __shfl_xor(s2, s, 32);
        }
        const float mean = s1 * (1.f / DIM);
        const float var  = s2 * (1.f / DIM) - mean * mean;
        const float rstd = rsqrtf(var + LN_EPS);

        if (n < N_NODES) {
            float* op = out + (size_t)n * DIM + lane * 8;
            #pragma unroll
            for (int i = 0; i < 8; ++i) {
                const int c = lane * 8 + i;
                op[i] = (y[i] - mean) * rstd * lng[c] + lnb[c];
            }
        }
    }
}

// ---------------- host launcher ----------------
extern "C" void kernel_launch(void* const* d_in, const int* in_sizes, int n_in,
                              void* d_out, int out_size, void* d_ws, size_t ws_size,
                              hipStream_t stream) {
    const float* feat  = (const float*)d_in[0];
    // d_in[1] Wl, d_in[2] bl, d_in[5] attn_l, d_in[6] attn_r: provably dead (see analysis)
    const float* Wr    = (const float*)d_in[3];
    const float* br    = (const float*)d_in[4];
    const float* rel_q = (const float*)d_in[7];
    const float* rel_k = (const float*)d_in[8];
    const float* lng   = (const float*)d_in[9];
    const float* lnb   = (const float*)d_in[10];
    const int*   dst0  = (const int*)d_in[12];
    const int*   dst1  = (const int*)d_in[14];
    const int*   dst2  = (const int*)d_in[16];
    const int E0 = in_sizes[12], E1 = in_sizes[14], E2 = in_sizes[16];

    // workspace: [ indicators: 3*N bytes | Wr^T bf16: 256*256*2 bytes ]
    uint8_t*        ind  = (uint8_t*)d_ws;
    unsigned short* WrTu = (unsigned short*)((char*)d_ws + ((3 * N_NODES + 255) & ~255));
    float* out = (float*)d_out;

    zero_ind_kernel<<<(3 * N_NODES + 255) / 256, 256, 0, stream>>>(ind, 3 * N_NODES);
    mark_kernel<<<(E0 + 255) / 256, 256, 0, stream>>>(dst0, ind,               E0);
    mark_kernel<<<(E1 + 255) / 256, 256, 0, stream>>>(dst1, ind + N_NODES,     E1);
    mark_kernel<<<(E2 + 255) / 256, 256, 0, stream>>>(dst2, ind + 2 * N_NODES, E2);
    wrT_kernel<<<DIM, DIM, 0, stream>>>(Wr, WrTu);
    latte_fused<<<(N_NODES + BM - 1) / BM, 256, 0, stream>>>(
        feat, WrTu, br, ind, rel_q, rel_k, lng, lnb, out);
}
